// MoELayer_50268297232581
// MI455X (gfx1250) — matrix operand order repaired and checked by reading, not compiled, not run on hardware
//
#include <hip/hip_runtime.h>
#include <hip/hip_bf16.h>

// Problem constants (from reference)
#define NB      4
#define NLOC    4096
#define NATOMS  (NB * NLOC)      // 16384
#define DIN     512
#define DOUT    512
#define NTYPES  128
#define TEBD    128
#define NE      8
#define TOPK    2
#define CAP     NATOMS           // worst case: every atom routes to one expert

typedef __attribute__((ext_vector_type(16))) __bf16 v16bf;
typedef __attribute__((ext_vector_type(8)))  __bf16 v8bf;
typedef __attribute__((ext_vector_type(8)))  float  v8f;

// ---------------------------------------------------------------- utilities
__device__ __forceinline__ float ftanh(float x) {
    x = fminf(10.0f, fmaxf(-10.0f, x));
    float t = __expf(2.0f * x);
    return (t - 1.0f) / (t + 1.0f);
}

// Native fp32 global atomic add (global_atomic_add_f32), avoiding the CAS-loop
// lowering of plain atomicAdd when -munsafe-fp-atomics is not set.
__device__ __forceinline__ void atomic_add_f32(float* p, float v) {
#if defined(__HIP_DEVICE_COMPILE__) && defined(__AMDGCN__)
    unsafeAtomicAdd(p, v);
#else
    atomicAdd(p, v);
#endif
}

// ---------------------------------------------------------------- kernel 0: zero output
__global__ void moe_zero_out(float* __restrict__ out, int n) {
    int i = blockIdx.x * blockDim.x + threadIdx.x;
    int i4 = i * 4;
    if (i4 + 3 < n) {
        *(float4*)(out + i4) = make_float4(0.f, 0.f, 0.f, 0.f);
    } else {
        for (int j = i4; j < n; ++j) out[j] = 0.f;
    }
}

// ---------------------------------------------------------------- kernel 1: routing (per type)
// logits = TE @ Wg  -> top-2 (lowest index wins ties) -> 2-way softmax
__global__ void moe_route(const float* __restrict__ te,   // [NTYPES, TEBD]
                          const float* __restrict__ Wg,   // [TEBD, NE]
                          float* __restrict__ typew,      // [NTYPES, 2]
                          int*   __restrict__ typei,      // [NTYPES, 2]
                          int*   __restrict__ cnt) {      // [NE]
    int t = threadIdx.x;
    if (t < NE) cnt[t] = 0;
    if (t >= NTYPES) return;

    float logits[NE];
#pragma unroll
    for (int e = 0; e < NE; ++e) logits[e] = 0.f;
    for (int d = 0; d < TEBD; ++d) {
        float v = te[t * TEBD + d];
#pragma unroll
        for (int e = 0; e < NE; ++e) logits[e] += v * Wg[d * NE + e];
    }
    int i1 = 0; float v1 = logits[0];
#pragma unroll
    for (int e = 1; e < NE; ++e)
        if (logits[e] > v1) { v1 = logits[e]; i1 = e; }
    int i2 = -1; float v2 = -3.4e38f;
#pragma unroll
    for (int e = 0; e < NE; ++e)
        if (e != i1 && logits[e] > v2) { v2 = logits[e]; i2 = e; }

    float z  = __expf(v2 - v1);          // v2 <= v1, no overflow
    float inv = 1.0f / (1.0f + z);
    typew[t * 2 + 0] = inv;
    typew[t * 2 + 1] = z * inv;
    typei[t * 2 + 0] = i1;
    typei[t * 2 + 1] = i2;
}

// ---------------------------------------------------------------- kernel 2: scatter atoms to experts
__global__ void moe_scatter(const int* __restrict__ atom_types,  // [NATOMS]
                            const float* __restrict__ typew,
                            const int*   __restrict__ typei,
                            int*   __restrict__ cnt,
                            int*   __restrict__ idxl,            // [NE, CAP]
                            float* __restrict__ wl) {            // [NE, CAP]
    int a = blockIdx.x * blockDim.x + threadIdx.x;
    if (a >= NATOMS) return;
    int t = atom_types[a];
#pragma unroll
    for (int k = 0; k < TOPK; ++k) {
        int e   = typei[t * 2 + k];
        float w = typew[t * 2 + k];
        int pos = atomicAdd(&cnt[e], 1);
        idxl[e * CAP + pos] = a;
        wl[e * CAP + pos]   = w;
    }
}

// ---------------------------------------------------------------- kernel 3: We [E,DIN,DOUT] f32 -> WeT [E,DOUT,DIN] bf16
__global__ void moe_transpose(const float* __restrict__ We, __bf16* __restrict__ WeT) {
    __shared__ float tile[32][33];
    int e  = blockIdx.z;
    int db = blockIdx.x * 32;     // DIN (K) base
    int ob = blockIdx.y * 32;     // DOUT (N) base
    int tx = threadIdx.x, ty = threadIdx.y;
    const float* src = We + (size_t)e * DIN * DOUT;
    __bf16* dst      = WeT + (size_t)e * DIN * DOUT;
#pragma unroll
    for (int i = 0; i < 32; i += 8)
        tile[ty + i][tx] = src[(size_t)(db + ty + i) * DOUT + ob + tx];
    __syncthreads();
#pragma unroll
    for (int i = 0; i < 32; i += 8)
        dst[(size_t)(ob + ty + i) * DIN + db + tx] = (__bf16)tile[tx][ty + i];
}

// ---------------------------------------------------------------- kernel 4: grouped GEMM + tanh + weighted combine
// grid: (2 col-segments of 256, 1024 row-tiles, 8 experts); block: 128 (4 waves)
// wave w handles col tiles [colseg*256 + w*64, +64) as 4 WMMA tiles of 16.
#define XPAD (DIN + 8)   // 1040 B row stride -> conflict-free b128 LDS reads
__global__ void moe_gemm(const float* __restrict__ x,      // [NATOMS, DIN]
                         const __bf16* __restrict__ WeT,   // [NE, DOUT, DIN]
                         const float* __restrict__ be,     // [NE, DOUT]
                         const int*   __restrict__ cnt,
                         const int*   __restrict__ idxl,
                         const float* __restrict__ wl,
                         float* __restrict__ out) {        // [NATOMS, DOUT]
    __shared__ __bf16 xt[16][XPAD];
    __shared__ float wrow[16];
    __shared__ int   ridx[16];

    const int e    = blockIdx.z;
    const int cntE = cnt[e];
    const int r0   = blockIdx.y * 16;
    if (r0 >= cntE) return;

    const int tid = threadIdx.x;
    if (tid < 16) {
        int rr  = r0 + tid;
        int src = rr < cntE ? rr : (cntE - 1);
        ridx[tid] = idxl[e * CAP + src];
        wrow[tid] = (rr < cntE) ? wl[e * CAP + src] : 0.0f;
    }
    __syncthreads();

    // gather 16 rows of x -> bf16 LDS tile (float4 coalesced)
    for (int i = tid; i < 16 * (DIN / 4); i += 128) {
        int row = i >> 7;              // DIN/4 = 128 float4 per row
        int c4  = i & 127;
        float4 v = *((const float4*)(x + (size_t)ridx[row] * DIN) + c4);
        __bf16* d = &xt[row][c4 * 4];
        d[0] = (__bf16)v.x; d[1] = (__bf16)v.y;
        d[2] = (__bf16)v.z; d[3] = (__bf16)v.w;
    }
    __syncthreads();

    const int lane = tid & 31;
    const int wv   = tid >> 5;
    const int lh   = lane >> 4;        // lane half
    const int ln   = lane & 15;
    const int colbase = blockIdx.x * 256 + wv * 64;

    const __bf16* wet_e = WeT + (size_t)e * DOUT * DIN;
    // B fragment row pointers: lane holds column N = ln, K = j + 16*lh (+k)
    const __bf16* brow0 = wet_e + (size_t)(colbase +  0 + ln) * DIN + 16 * lh;
    const __bf16* brow1 = wet_e + (size_t)(colbase + 16 + ln) * DIN + 16 * lh;
    const __bf16* brow2 = wet_e + (size_t)(colbase + 32 + ln) * DIN + 16 * lh;
    const __bf16* brow3 = wet_e + (size_t)(colbase + 48 + ln) * DIN + 16 * lh;
    // A fragment: lane holds row M = ln, K = k+16*lh .. k+16*lh+15
    const __bf16* arow = &xt[ln][16 * lh];

    v8f acc0 = {}, acc1 = {}, acc2 = {}, acc3 = {};
#pragma unroll
    for (int k = 0; k < DIN; k += 32) {
        v8bf alo = *(const v8bf*)(arow + k);
        v8bf ahi = *(const v8bf*)(arow + k + 8);
        v16bf a = __builtin_shufflevector(alo, ahi,
                    0,1,2,3,4,5,6,7,8,9,10,11,12,13,14,15);
        v16bf b0 = *(const v16bf*)(brow0 + k);
        v16bf b1 = *(const v16bf*)(brow1 + k);
        v16bf b2 = *(const v16bf*)(brow2 + k);
        v16bf b3 = *(const v16bf*)(brow3 + k);
        acc0 = __builtin_amdgcn_wmma_f32_16x16x32_bf16(false, a, false, b0, (short)0, acc0, false, false);
        acc1 = __builtin_amdgcn_wmma_f32_16x16x32_bf16(false, a, false, b1, (short)0, acc1, false, false);
        acc2 = __builtin_amdgcn_wmma_f32_16x16x32_bf16(false, a, false, b2, (short)0, acc2, false, false);
        acc3 = __builtin_amdgcn_wmma_f32_16x16x32_bf16(false, a, false, b3, (short)0, acc3, false, false);
    }

    // epilogue: + be, tanh, per-row weight, native-atomic combine
    // (each atom receives exactly 2 expert contributions)
    const float* beE = be + e * DOUT;
    v8f accs[4] = {acc0, acc1, acc2, acc3};
#pragma unroll
    for (int t = 0; t < 4; ++t) {
        int col = colbase + 16 * t + ln;
        float bias = beE[col];
#pragma unroll
        for (int r = 0; r < 8; ++r) {
            int M = r + 8 * lh;
            float val = ftanh(accs[t][r] + bias) * wrow[M];
            atomic_add_f32(&out[(size_t)ridx[M] * DOUT + col], val);
        }
    }
}

// ---------------------------------------------------------------- host entry
extern "C" void kernel_launch(void* const* d_in, const int* in_sizes, int n_in,
                              void* d_out, int out_size, void* d_ws, size_t ws_size,
                              hipStream_t stream) {
    const float* x   = (const float*)d_in[0];
    const float* te  = (const float*)d_in[1];
    const int*   at  = (const int*)  d_in[2];
    const float* Wg  = (const float*)d_in[3];
    const float* We  = (const float*)d_in[4];
    const float* be  = (const float*)d_in[5];
    float* out = (float*)d_out;

    // workspace layout (~5.25 MB)
    char* ws = (char*)d_ws;
    float* typew = (float*)(ws + 0);                            // 1 KB
    int*   typei = (int*)  (ws + 1024);                         // 1 KB
    int*   cnt   = (int*)  (ws + 2048);                         // 32 B (padded)
    int*   idxl  = (int*)  (ws + 4096);                         // 512 KB
    float* wl    = (float*)(ws + 4096 + (size_t)NE * CAP * 4);  // 512 KB
    __bf16* WeT  = (__bf16*)(ws + 4096 + 2 * (size_t)NE * CAP * 4); // 4 MB

    // 1. zero output (harness poisons it; atomics accumulate into it)
    {
        int n4 = (out_size + 3) / 4;
        moe_zero_out<<<(n4 + 255) / 256, 256, 0, stream>>>(out, out_size);
    }
    // 2. routing table per type (+ zero expert counters)
    moe_route<<<1, 128, 0, stream>>>(te, Wg, typew, typei, cnt);
    // 3. scatter atoms into per-expert lists
    moe_scatter<<<NATOMS / 256, 256, 0, stream>>>(at, typew, typei, cnt, idxl, wl);
    // 4. We -> WeT (bf16, [E, DOUT, DIN])
    moe_transpose<<<dim3(DIN / 32, DOUT / 32, NE), dim3(32, 8), 0, stream>>>(We, WeT);
    // 5. grouped GEMM over per-expert row tiles (dead tiles exit on cnt check)
    moe_gemm<<<dim3(2, CAP / 16, NE), 128, 0, stream>>>(x, WeT, be, cnt, idxl, wl, out);
}
